// ALIGNNAtomWise_47425028883060
// MI455X (gfx1250) — matrix operand order repaired and checked by reading, not compiled, not run on hardware
//
#include <hip/hip_runtime.h>
#include <hip/hip_bf16.h>

// ---------------------------------------------------------------------------
// ALIGNN forward for MI455X (gfx1250, wave32, WMMA).
// All dense GEMMs (K=256/96/64) run through v_wmma_f32_16x16x32_f16 with
// f16 weights pre-transposed to [N,Kpad] for contiguous B-fragment loads.
// ---------------------------------------------------------------------------

#define NN_  4096
#define EE_  49152
#define TT_  262144
#define HD   256
#define AINK 92
#define EBK  80
#define TBK  40
#define EMBD 64

typedef __attribute__((ext_vector_type(16))) _Float16 v16h;
typedef __attribute__((ext_vector_type(8)))  _Float16 v8h;
typedef __attribute__((ext_vector_type(8)))  float    v8f;

// ---------------------------------------------------------------------------
// Weight conversion: W[K,N] f32 -> Wt[N,Kpad] f16 (transposed + zero padded).
// ---------------------------------------------------------------------------
__global__ void k_wconv(const float* __restrict__ W, _Float16* __restrict__ O,
                        int K, int N, int Kpad, int nmats,
                        long inStride, long outStride) {
  long total = (long)nmats * N * Kpad;
  for (long i = blockIdx.x * (long)blockDim.x + threadIdx.x; i < total;
       i += (long)gridDim.x * blockDim.x) {
    long per = (long)N * Kpad;
    int  m   = (int)(i / per);
    long rem = i - (long)m * per;
    int  n   = (int)(rem / Kpad);
    int  k   = (int)(rem - (long)n * Kpad);
    float v  = (k < K) ? W[(long)m * inStride + (long)k * N + n] : 0.0f;
    O[(long)m * outStride + (long)n * Kpad + k] = (_Float16)v;
  }
}

// ---------------------------------------------------------------------------
// WMMA GEMM: Y[M,NOUT] = X[M,Kact] @ W + bias.  Wt is [NOUT,KPAD] f16.
// Block = 256 threads = 8 waves; block tile = 64 rows x NOUT cols.
// Wave (rs,half): row strip rs (16 rows), column tiles t = half + 2q.
// A fragment: lane row m, K = kb + 16*(lane>>4) + j (two 16B LDS loads).
// B fragment: lane column n, K = kb + 16*(lane>>4) + j,
// contiguous 32B from the transposed weight matrix.
// ---------------------------------------------------------------------------
template <int KPAD, int NOUT>
__global__ __launch_bounds__(256) void k_gemm(const float* __restrict__ X,
                                              const _Float16* __restrict__ Wt,
                                              const float* __restrict__ bias,
                                              float* __restrict__ Y,
                                              int M, int Kact) {
  constexpr int LDA = KPAD + 8;          // pad rows: 528B pitch stays 16B aligned
  __shared__ __align__(16) _Float16 As[64 * LDA];
  const int tid  = threadIdx.x;
  const int row0 = blockIdx.x * 64;

  // Cooperative f32 -> f16 stage of the 64-row A tile (zero padded).
  for (int i = tid; i < 64 * KPAD; i += 256) {
    const int r  = i / KPAD;
    const int c  = i - r * KPAD;
    const int gr = row0 + r;
    float v = 0.0f;
    if (gr < M && c < Kact) v = X[(size_t)gr * Kact + c];
    As[r * LDA + c] = (_Float16)v;
  }
  __syncthreads();

  const int lane = tid & 31;
  const int wave = tid >> 5;
  const int lh   = lane >> 4;
  const int ll   = lane & 15;
  const int rs   = wave >> 1;            // row strip 0..3
  const int half = wave & 1;             // column parity
  constexpr int NT = NOUT / 16;
  constexpr int QT = NT / 2;             // col tiles per wave
  constexpr int KS = KPAD / 32;

  v8f acc[QT] = {};

  for (int ks = 0; ks < KS; ++ks) {
    const int kb = ks << 5;
    if (ks + 1 < KS)                     // L2 prefetch of next K slab of weights
      __builtin_prefetch(Wt + (size_t)(half * 16 + ll) * KPAD + kb + 32, 0, 0);

    union { v16h v; v8h h[2]; } au;
    const _Float16* ap = &As[(rs * 16 + ll) * LDA + kb + 16 * lh];
    au.h[0] = *(const v8h*)ap;           // K = kb + 16*lh + [0..7]
    au.h[1] = *(const v8h*)(ap + 8);     // K = kb + 16*lh + 8 + [0..7]
    const v16h a = au.v;

#pragma unroll
    for (int q = 0; q < QT; ++q) {
      const int n = (half + 2 * q) * 16 + ll;
      const v16h b = *(const v16h*)(Wt + (size_t)n * KPAD + kb + 16 * lh);
      acc[q] = __builtin_amdgcn_wmma_f32_16x16x32_f16(
          false, a, false, b, (short)0, acc[q], false, false);
    }
  }

#pragma unroll
  for (int q = 0; q < QT; ++q) {
    const int n  = (half + 2 * q) * 16 + ll;
    const float bv = bias[n];
#pragma unroll
    for (int v = 0; v < 8; ++v) {
      const int r = row0 + rs * 16 + v + 8 * lh;   // C/D layout: M = v + 8*lh
      if (r < M) Y[(size_t)r * NOUT + n] = acc[q][v] + bv;
    }
  }
}

// ---------------------------------------------------------------------------
// RBF kernels.
// ---------------------------------------------------------------------------
__global__ void k_rbf_edge(const float* __restrict__ r, float* __restrict__ out,
                           int E) {
  int i = blockIdx.x * blockDim.x + threadIdx.x;
  if (i >= E) return;
  float x = r[3 * i], y = r[3 * i + 1], z = r[3 * i + 2];
  float L = sqrtf(x * x + y * y + z * z);
  const float step  = 8.0f / (EBK - 1);
  const float gamma = 1.0f / step;
  for (int b = 0; b < EBK; ++b) {
    float d = L - step * b;
    out[(size_t)i * EBK + b] = __expf(-gamma * d * d);
  }
}

__global__ void k_rbf_ang(const float* __restrict__ r, const int* __restrict__ ls,
                          const int* __restrict__ ld, float* __restrict__ out,
                          int T) {
  int i = blockIdx.x * blockDim.x + threadIdx.x;
  if (i >= T) return;
  int a = ls[i], b = ld[i];
  float ax = -r[3 * a], ay = -r[3 * a + 1], az = -r[3 * a + 2];
  float bx =  r[3 * b], by =  r[3 * b + 1], bz =  r[3 * b + 2];
  float dt = ax * bx + ay * by + az * bz;
  float na = sqrtf(ax * ax + ay * ay + az * az);
  float nb = sqrtf(bx * bx + by * by + bz * bz);
  float c  = dt / (na * nb);
  c = fminf(1.0f, fmaxf(-1.0f, c));
  const float step  = 2.0f / (TBK - 1);
  const float gamma = 1.0f / step;
  for (int bn = 0; bn < TBK; ++bn) {
    float d = c - (-1.0f + step * bn);
    out[(size_t)i * TBK + bn] = __expf(-gamma * d * d);
  }
}

// ---------------------------------------------------------------------------
// Wave-per-row LayerNorm + SiLU (in place).  wave32 shuffle reductions.
// ---------------------------------------------------------------------------
template <int W>
__global__ __launch_bounds__(256) void k_ln_silu(float* __restrict__ Y,
                                                 const float* __restrict__ s,
                                                 const float* __restrict__ b,
                                                 int M) {
  const int wave = threadIdx.x >> 5, lane = threadIdx.x & 31;
  const int row  = blockIdx.x * 8 + wave;
  if (row >= M) return;
  constexpr int E = W / 32;
  float t[E];
  float sum = 0.0f;
#pragma unroll
  for (int j = 0; j < E; ++j) {
    t[j] = Y[(size_t)row * W + j * 32 + lane];
    sum += t[j];
  }
  for (int o = 16; o; o >>= 1) sum += __shfl_xor(sum, o, 32);
  const float mu = sum / W;
  float vs = 0.0f;
#pragma unroll
  for (int j = 0; j < E; ++j) { float d = t[j] - mu; vs += d * d; }
  for (int o = 16; o; o >>= 1) vs += __shfl_xor(vs, o, 32);
  const float inv = rsqrtf(vs / W + 1e-5f);
#pragma unroll
  for (int j = 0; j < E; ++j) {
    const int c = j * 32 + lane;
    float u  = (t[j] - mu) * inv * s[c] + b[c];
    float sg = 1.0f / (1.0f + __expf(-u));
    Y[(size_t)row * W + c] = u * sg;
  }
}

// ---------------------------------------------------------------------------
// EGGC helpers.
// ---------------------------------------------------------------------------
__global__ void k_zero(float* __restrict__ p, long n) {
  long i = blockIdx.x * (long)blockDim.x + threadIdx.x;
  if (i < n) p[i] = 0.0f;
}

// m[e,:] += g0[src[e],:] + g1[dst[e],:]   (gate pre-activation, in place)
__global__ void k_gather_add(float* __restrict__ m, const float* __restrict__ g0,
                             const float* __restrict__ g1,
                             const int* __restrict__ src,
                             const int* __restrict__ dst, long total) {
  long i = blockIdx.x * (long)blockDim.x + threadIdx.x;
  if (i >= total) return;
  long e = i >> 8;
  int  c = (int)(i & 255);
  int  s = src[e], d = dst[e];
  m[i] += g0[(size_t)s * HD + c] + g1[(size_t)d * HD + c];
}

// sigma = sigmoid(m); ssh[dst] += xw4[src]*sigma; ss[dst] += sigma
__global__ void k_sigma_scatter(const float* __restrict__ m,
                                const float* __restrict__ xw4,
                                const int* __restrict__ src,
                                const int* __restrict__ dst,
                                float* __restrict__ ssh, float* __restrict__ ss,
                                long total) {
  long i = blockIdx.x * (long)blockDim.x + threadIdx.x;
  if (i >= total) return;
  long e = i >> 8;
  int  c = (int)(i & 255);
  float sg = 1.0f / (1.0f + __expf(-m[i]));
  int s = src[e], d = dst[e];
  __hip_atomic_fetch_add(&ssh[(size_t)d * HD + c], xw4[(size_t)s * HD + c] * sg,
                         __ATOMIC_RELAXED, __HIP_MEMORY_SCOPE_AGENT);
  __hip_atomic_fetch_add(&ss[(size_t)d * HD + c], sg,
                         __ATOMIC_RELAXED, __HIP_MEMORY_SCOPE_AGENT);
}

// x += silu(ln(xw3 + ssh/(ss+1e-6)))
__global__ __launch_bounds__(256) void k_node_update(
    float* __restrict__ x, const float* __restrict__ xw3,
    const float* __restrict__ ssh, const float* __restrict__ ss,
    const float* __restrict__ lns, const float* __restrict__ lnb, int M) {
  const int wave = threadIdx.x >> 5, lane = threadIdx.x & 31;
  const int row  = blockIdx.x * 8 + wave;
  if (row >= M) return;
  float t[8];
  float sum = 0.0f;
#pragma unroll
  for (int j = 0; j < 8; ++j) {
    const size_t o = (size_t)row * HD + j * 32 + lane;
    t[j] = xw3[o] + ssh[o] / (ss[o] + 1e-6f);
    sum += t[j];
  }
  for (int o = 16; o; o >>= 1) sum += __shfl_xor(sum, o, 32);
  const float mu = sum / HD;
  float vs = 0.0f;
#pragma unroll
  for (int j = 0; j < 8; ++j) { float d = t[j] - mu; vs += d * d; }
  for (int o = 16; o; o >>= 1) vs += __shfl_xor(vs, o, 32);
  const float inv = rsqrtf(vs / HD + 1e-5f);
#pragma unroll
  for (int j = 0; j < 8; ++j) {
    const int c = j * 32 + lane;
    float u  = (t[j] - mu) * inv * lns[c] + lnb[c];
    float sg = 1.0f / (1.0f + __expf(-u));
    x[(size_t)row * HD + c] += u * sg;
  }
}

// e += silu(ln(m))
__global__ __launch_bounds__(256) void k_edge_update(
    float* __restrict__ e, const float* __restrict__ m,
    const float* __restrict__ lns, const float* __restrict__ lnb, int M) {
  const int wave = threadIdx.x >> 5, lane = threadIdx.x & 31;
  const int row  = blockIdx.x * 8 + wave;
  if (row >= M) return;
  float t[8];
  float sum = 0.0f;
#pragma unroll
  for (int j = 0; j < 8; ++j) {
    t[j] = m[(size_t)row * HD + j * 32 + lane];
    sum += t[j];
  }
  for (int o = 16; o; o >>= 1) sum += __shfl_xor(sum, o, 32);
  const float mu = sum / HD;
  float vs = 0.0f;
#pragma unroll
  for (int j = 0; j < 8; ++j) { float d = t[j] - mu; vs += d * d; }
  for (int o = 16; o; o >>= 1) vs += __shfl_xor(vs, o, 32);
  const float inv = rsqrtf(vs / HD + 1e-5f);
#pragma unroll
  for (int j = 0; j < 8; ++j) {
    const int c = j * 32 + lane;
    float u  = (t[j] - mu) * inv * lns[c] + lnb[c];
    float sg = 1.0f / (1.0f + __expf(-u));
    e[(size_t)row * HD + c] += u * sg;
  }
}

// atomwise = x @ fc_W + fc_b  (wave-per-row dot product)
__global__ __launch_bounds__(256) void k_fc(const float* __restrict__ x,
                                            const float* __restrict__ w,
                                            const float* __restrict__ fb,
                                            float* __restrict__ out, int M) {
  const int wave = threadIdx.x >> 5, lane = threadIdx.x & 31;
  const int row  = blockIdx.x * 8 + wave;
  if (row >= M) return;
  float sum = 0.0f;
#pragma unroll
  for (int j = 0; j < 8; ++j) {
    const int c = j * 32 + lane;
    sum += x[(size_t)row * HD + c] * w[c];
  }
  for (int o = 16; o; o >>= 1) sum += __shfl_xor(sum, o, 32);
  if (lane == 0) out[row] = sum + fb[0];
}

__global__ __launch_bounds__(256) void k_mean(const float* __restrict__ a,
                                              float* __restrict__ out0, int M) {
  __shared__ float red[256];
  float s = 0.0f;
  for (int i = threadIdx.x; i < M; i += 256) s += a[i];
  red[threadIdx.x] = s;
  __syncthreads();
  for (int o = 128; o; o >>= 1) {
    if ((int)threadIdx.x < o) red[threadIdx.x] += red[threadIdx.x + o];
    __syncthreads();
  }
  if (threadIdx.x == 0) out0[0] = red[0] / (float)M;
}

// ---------------------------------------------------------------------------
// Host orchestration.
// ---------------------------------------------------------------------------
static void run_eggc(float* x, float* e, const int* sp, const int* dp,
                     int NNn, int NEe, const _Float16* wt, const float* bb,
                     const float* lns, const float* lnb, float* g0, float* g1,
                     float* xw3, float* xw4, float* m, hipStream_t st) {
  const int gbn = (NNn + 63) / 64;
  const int gbe = (NEe + 63) / 64;
  k_gemm<HD, HD><<<gbn, 256, 0, st>>>(x, wt + 0 * HD * HD, bb + 0 * HD, g0, NNn, HD);
  k_gemm<HD, HD><<<gbn, 256, 0, st>>>(x, wt + 1 * HD * HD, bb + 1 * HD, g1, NNn, HD);
  k_gemm<HD, HD><<<gbe, 256, 0, st>>>(e, wt + 2 * HD * HD, bb + 2 * HD, m, NEe, HD);
  k_gemm<HD, HD><<<gbn, 256, 0, st>>>(x, wt + 3 * HD * HD, bb + 3 * HD, xw3, NNn, HD);
  k_gemm<HD, HD><<<gbn, 256, 0, st>>>(x, wt + 4 * HD * HD, bb + 4 * HD, xw4, NNn, HD);

  const long tote = (long)NEe * HD;
  const int  gte  = (int)((tote + 255) / 256);
  k_gather_add<<<gte, 256, 0, st>>>(m, g0, g1, sp, dp, tote);

  const long totn = (long)NNn * HD;
  const int  gtn  = (int)((totn + 255) / 256);
  k_zero<<<gtn, 256, 0, st>>>(g0, totn);   // g0/g1 reused as segment sums
  k_zero<<<gtn, 256, 0, st>>>(g1, totn);
  k_sigma_scatter<<<gte, 256, 0, st>>>(m, xw4, sp, dp, g0, g1, tote);

  k_node_update<<<(NNn + 7) / 8, 256, 0, st>>>(x, xw3, g0, g1, lns, lnb, NNn);
  k_edge_update<<<(NEe + 7) / 8, 256, 0, st>>>(e, m, lns + HD, lnb + HD, NEe);
}

extern "C" void kernel_launch(void* const* d_in, const int* in_sizes, int n_in,
                              void* d_out, int out_size, void* d_ws,
                              size_t ws_size, hipStream_t stream) {
  (void)in_sizes; (void)n_in; (void)out_size; (void)ws_size;

  const float* atom_features = (const float*)d_in[0];
  const float* rvec          = (const float*)d_in[1];
  const int*   src           = (const int*)d_in[2];
  const int*   dst           = (const int*)d_in[3];
  const int*   lsrc          = (const int*)d_in[4];
  const int*   ldst          = (const int*)d_in[5];
  const float* atom_W    = (const float*)d_in[6];
  const float* atom_b    = (const float*)d_in[7];
  const float* atom_ln_s = (const float*)d_in[8];
  const float* atom_ln_b = (const float*)d_in[9];
  const float* edge_W1   = (const float*)d_in[10];
  const float* edge_b1   = (const float*)d_in[11];
  const float* edge_ln1s = (const float*)d_in[12];
  const float* edge_ln1b = (const float*)d_in[13];
  const float* edge_W2   = (const float*)d_in[14];
  const float* edge_b2   = (const float*)d_in[15];
  const float* edge_ln2s = (const float*)d_in[16];
  const float* edge_ln2b = (const float*)d_in[17];
  const float* ang_W1    = (const float*)d_in[18];
  const float* ang_b1    = (const float*)d_in[19];
  const float* ang_ln1s  = (const float*)d_in[20];
  const float* ang_ln1b  = (const float*)d_in[21];
  const float* ang_W2    = (const float*)d_in[22];
  const float* ang_b2    = (const float*)d_in[23];
  const float* ang_ln2s  = (const float*)d_in[24];
  const float* ang_ln2b  = (const float*)d_in[25];
  const float* eggc_W    = (const float*)d_in[26];
  const float* eggc_b    = (const float*)d_in[27];
  const float* eggc_ln_s = (const float*)d_in[28];
  const float* eggc_ln_b = (const float*)d_in[29];
  const float* fc_W      = (const float*)d_in[30];
  const float* fc_b      = (const float*)d_in[31];
  float* out = (float*)d_out;

  // -------- workspace layout (f32 region then f16 weight region) ----------
  float* wsf = (float*)d_ws;
  size_t o = 0;
  float* X   = wsf + o; o += (size_t)NN_ * HD;
  float* Yb  = wsf + o; o += (size_t)EE_ * HD;
  float* Zb  = wsf + o; o += (size_t)TT_ * HD;
  float* Mb  = wsf + o; o += (size_t)TT_ * HD;   // gate buffer (also RBF/EMB temp)
  float* G0  = wsf + o; o += (size_t)EE_ * HD;
  float* G1  = wsf + o; o += (size_t)EE_ * HD;
  float* XW3 = wsf + o; o += (size_t)EE_ * HD;
  float* XW4 = wsf + o; o += (size_t)EE_ * HD;
  _Float16* WT      = (_Float16*)(wsf + o);
  _Float16* wt_eggc = WT;                                   // 60 x 256x256
  _Float16* wt_atom = wt_eggc + (size_t)60 * HD * HD;       // 256 x 96
  _Float16* wt_e1   = wt_atom + 256 * 96;                   // 64 x 96
  _Float16* wt_e2   = wt_e1 + 64 * 96;                      // 256 x 64
  _Float16* wt_a1   = wt_e2 + 256 * 64;                     // 64 x 64
  _Float16* wt_a2   = wt_a1 + 64 * 64;                      // 256 x 64

  float* RBF  = Mb;                          // [T,40] / [E,80]
  float* EMBb = Mb + (size_t)TT_ * TBK;      // [T,64] / [E,64]

  // -------- 1) convert weights: f32 [K,N] -> f16 [N,Kpad] transposed -------
  {
    long tot = 60L * HD * HD;
    k_wconv<<<(int)((tot + 255) / 256), 256, 0, stream>>>(
        eggc_W, wt_eggc, HD, HD, HD, 60, (long)HD * HD, (long)HD * HD);
  }
  k_wconv<<<(256 * 96 + 255) / 256, 256, 0, stream>>>(atom_W, wt_atom, AINK, HD, 96, 1, 0, 0);
  k_wconv<<<(64 * 96 + 255) / 256, 256, 0, stream>>>(edge_W1, wt_e1, EBK, EMBD, 96, 1, 0, 0);
  k_wconv<<<(256 * 64 + 255) / 256, 256, 0, stream>>>(edge_W2, wt_e2, EMBD, HD, 64, 1, 0, 0);
  k_wconv<<<(64 * 64 + 255) / 256, 256, 0, stream>>>(ang_W1, wt_a1, TBK, EMBD, 64, 1, 0, 0);
  k_wconv<<<(256 * 64 + 255) / 256, 256, 0, stream>>>(ang_W2, wt_a2, EMBD, HD, 64, 1, 0, 0);

  // -------- 2) angle (line-graph) embedding: z ----------------------------
  k_rbf_ang<<<(TT_ + 255) / 256, 256, 0, stream>>>(rvec, lsrc, ldst, RBF, TT_);
  k_gemm<64, 64><<<TT_ / 64, 256, 0, stream>>>(RBF, wt_a1, ang_b1, EMBb, TT_, TBK);
  k_ln_silu<64><<<TT_ / 8, 256, 0, stream>>>(EMBb, ang_ln1s, ang_ln1b, TT_);
  k_gemm<64, HD><<<TT_ / 64, 256, 0, stream>>>(EMBb, wt_a2, ang_b2, Zb, TT_, EMBD);
  k_ln_silu<HD><<<TT_ / 8, 256, 0, stream>>>(Zb, ang_ln2s, ang_ln2b, TT_);

  // -------- 3) bond embedding: y ------------------------------------------
  k_rbf_edge<<<(EE_ + 255) / 256, 256, 0, stream>>>(rvec, RBF, EE_);
  k_gemm<96, 64><<<EE_ / 64, 256, 0, stream>>>(RBF, wt_e1, edge_b1, EMBb, EE_, EBK);
  k_ln_silu<64><<<EE_ / 8, 256, 0, stream>>>(EMBb, edge_ln1s, edge_ln1b, EE_);
  k_gemm<64, HD><<<EE_ / 64, 256, 0, stream>>>(EMBb, wt_e2, edge_b2, Yb, EE_, EMBD);
  k_ln_silu<HD><<<EE_ / 8, 256, 0, stream>>>(Yb, edge_ln2s, edge_ln2b, EE_);

  // -------- 4) atom embedding: x ------------------------------------------
  k_gemm<96, HD><<<NN_ / 64, 256, 0, stream>>>(atom_features, wt_atom, atom_b, X, NN_, AINK);
  k_ln_silu<HD><<<NN_ / 8, 256, 0, stream>>>(X, atom_ln_s, atom_ln_b, NN_);

  // -------- 5) 4 ALIGNN pairs + 4 GCN layers ------------------------------
  for (int i = 0; i < 4; ++i) {
    const int b0 = 2 * i, b1 = 2 * i + 1;
    run_eggc(X, Yb, src, dst, NN_, EE_,
             wt_eggc + (size_t)b0 * 5 * HD * HD, eggc_b + (size_t)b0 * 5 * HD,
             eggc_ln_s + (size_t)b0 * 2 * HD, eggc_ln_b + (size_t)b0 * 2 * HD,
             G0, G1, XW3, XW4, Mb, stream);
    run_eggc(Yb, Zb, lsrc, ldst, EE_, TT_,
             wt_eggc + (size_t)b1 * 5 * HD * HD, eggc_b + (size_t)b1 * 5 * HD,
             eggc_ln_s + (size_t)b1 * 2 * HD, eggc_ln_b + (size_t)b1 * 2 * HD,
             G0, G1, XW3, XW4, Mb, stream);
  }
  for (int i = 0; i < 4; ++i) {
    const int b = 8 + i;
    run_eggc(X, Yb, src, dst, NN_, EE_,
             wt_eggc + (size_t)b * 5 * HD * HD, eggc_b + (size_t)b * 5 * HD,
             eggc_ln_s + (size_t)b * 2 * HD, eggc_ln_b + (size_t)b * 2 * HD,
             G0, G1, XW3, XW4, Mb, stream);
  }

  // -------- 6) readout: atomwise energies + mean --------------------------
  k_fc<<<(NN_ + 7) / 8, 256, 0, stream>>>(X, fc_W, fc_b, out + 1, NN_);
  k_mean<<<1, 256, 0, stream>>>(out + 1, out, NN_);
}
